// MAMBA_58428735095306
// MI455X (gfx1250) — matrix operand order repaired and checked by reading, not compiled, not run on hardware
//
#include <hip/hip_runtime.h>
#include <hip/hip_bf16.h>

#define NLAYER 4
#define DMOD   128
#define DINNER 256
#define DST    16
#define DTRANK 8
#define NCLS   256
#define NB     24
#define LSEQ   3072
#define TSEQ   1024
#define GDBL   40      // DTRANK + 2*DST
#define NROWS  (NB*TSEQ)

typedef __attribute__((ext_vector_type(16))) __bf16          v16bf;
typedef __attribute__((ext_vector_type(8)))  float           v8f;
typedef __attribute__((ext_vector_type(8)))  unsigned short  u16x8;
typedef __attribute__((ext_vector_type(4)))  unsigned int    v4u;
typedef __attribute__((ext_vector_type(8)))  int             v8i_;
typedef __attribute__((ext_vector_type(4)))  int             v4i_;

union BFrag { u16x8 h[2]; v16bf v; };

__device__ __forceinline__ float bf2f(unsigned short u) {
  unsigned int x = ((unsigned int)u) << 16;
  return __builtin_bit_cast(float, x);
}
__device__ __forceinline__ unsigned short f2bf(float f) {
  unsigned int x = __builtin_bit_cast(unsigned int, f);
  unsigned int r = (x + 0x7FFFu + ((x >> 16) & 1u)) >> 16;
  return (unsigned short)r;
}

// ---- Tensor Data Mover: 2D bf16 tile (tile0 x tile1) from global to LDS ----
// D# per CDNA5 ISA. data_size=2B. OOB rows/cols read as 0.
// amdgpu-toolchain (clang-23) 6-arg builtin form.
__device__ __forceinline__ void tdm_load_2d(unsigned ldsOff,
                                            const unsigned short* gptr,
                                            int dim0, int dim1,
                                            int tile0, int tile1, int stride0) {
  unsigned long long ga = (unsigned long long)gptr;
  v4u g0;
  g0[0] = 1u;                                              // count=1, no gather
  g0[1] = ldsOff;                                          // lds_addr (bytes)
  g0[2] = (unsigned)(ga & 0xFFFFFFFFull);                  // global_addr lo
  g0[3] = (unsigned)((ga >> 32) & 0x01FFFFFFull) | (2u << 30); // hi + type=2
  v8i_ g1;
  g1[0] = (1 << 16);                                       // data_size = 2 bytes
  g1[1] = (int)(((unsigned)dim0 & 0xFFFFu) << 16);         // tensor_dim0[15:0]
  g1[2] = (int)((((unsigned)dim0) >> 16) | (((unsigned)dim1 & 0xFFFFu) << 16));
  g1[3] = (int)((((unsigned)dim1) >> 16) | (((unsigned)tile0) << 16)); // tile_dim0
  g1[4] = tile1;                                           // tile_dim1 (tile_dim2=0)
  g1[5] = stride0;                                         // tensor_dim0_stride lo
  g1[6] = 0;
  g1[7] = 0;
  v4i_ z4 = {0, 0, 0, 0};
  v8i_ z8 = {0, 0, 0, 0, 0, 0, 0, 0};
  __builtin_amdgcn_tensor_load_to_lds(g0, g1, z4, z4, z8, 0);
}

// ---------------------------------------------------------------- stem conv
__global__ __launch_bounds__(128)
void k_conv0(const float* __restrict__ x, const float* __restrict__ cw,
             const float* __restrict__ cb, float* __restrict__ h0) {
  int bt = blockIdx.x;                 // b*T + t
  int b = bt / TSEQ, t = bt % TSEQ;
  int d = threadIdx.x;
  float acc = cb[d];
  #pragma unroll
  for (int k = 0; k < 3; k++) {
    int xi = 3 * t + k - 1;
    if (xi >= 0 && xi < LSEQ) acc += x[(size_t)b * LSEQ + xi] * cw[d * 3 + k];
  }
  h0[(size_t)bt * DMOD + d] = acc;
}

// ------------------------------------------------------------- f32 -> bf16
__global__ void k_cvt_bf16(const float* __restrict__ src,
                           unsigned short* __restrict__ dst, int n) {
  int i = blockIdx.x * blockDim.x + threadIdx.x;
  for (; i < n; i += gridDim.x * blockDim.x) dst[i] = f2bf(src[i]);
}

// ------------------------------------------------------- (reversed) copy
__global__ void k_copy_rev(const float* __restrict__ src, float* __restrict__ dst,
                           int rev) {
  size_t i = (size_t)blockIdx.x * blockDim.x + threadIdx.x;
  size_t total = (size_t)NROWS * DMOD;
  if (i >= total) return;
  int d = (int)(i % DMOD);
  size_t bt = i / DMOD;
  int t = (int)(bt % TSEQ); int b = (int)(bt / TSEQ);
  int ts = rev ? (TSEQ - 1 - t) : t;
  dst[i] = src[((size_t)b * TSEQ + ts) * DMOD + d];
}

// --------------------------------------------- residual accumulate + LN
// one wave (32 lanes) per row of DMOD=128
__global__ __launch_bounds__(256)
void k_ln_res(const float* __restrict__ h, float* __restrict__ resid,
              const float* __restrict__ w, const float* __restrict__ bia,
              unsigned short* __restrict__ out, int first) {
  int row  = blockIdx.x * 8 + (threadIdx.x >> 5);
  int lane = threadIdx.x & 31;
  float r[4]; float s = 0.f;
  #pragma unroll
  for (int j = 0; j < 4; j++) {
    int c = lane + 32 * j;
    float v = h[(size_t)row * DMOD + c];
    if (!first) v += resid[(size_t)row * DMOD + c];
    resid[(size_t)row * DMOD + c] = v;
    r[j] = v; s += v;
  }
  #pragma unroll
  for (int o = 16; o > 0; o >>= 1) s += __shfl_xor(s, o, 32);
  float mu = s * (1.f / DMOD);
  float vs = 0.f;
  #pragma unroll
  for (int j = 0; j < 4; j++) { float d = r[j] - mu; vs += d * d; }
  #pragma unroll
  for (int o = 16; o > 0; o >>= 1) vs += __shfl_xor(vs, o, 32);
  float rstd = rsqrtf(vs * (1.f / DMOD) + 1e-5f);
  #pragma unroll
  for (int j = 0; j < 4; j++) {
    int c = lane + 32 * j;
    out[(size_t)row * DMOD + c] = f2bf((r[j] - mu) * rstd * w[c] + bia[c]);
  }
}

// ----------------------------------------------------------- WMMA GEMM
// C[M,N] = A[M,K](bf16) * W[N,K](bf16)^T
// 128x128 block tile, 8 waves, each wave 32x64 = 2x4 WMMA fragments.
// Tiles staged to LDS by the Tensor Data Mover (waves 0/1), TENSORcnt-tracked.
#define BM 128
#define BN 128
#define BK 32
__global__ __launch_bounds__(256)
void k_gemm_bf16(const unsigned short* __restrict__ A,
                 const unsigned short* __restrict__ W,
                 float* __restrict__ Cf, unsigned short* __restrict__ Cb,
                 int M, int N, int K) {
  __shared__ __align__(16) unsigned short As[BM][BK];
  __shared__ __align__(16) unsigned short Ws[BN][BK];
  const int tid  = threadIdx.x;
  const int lane = tid & 31;
  const int wave = tid >> 5;
  const int wr = wave >> 1;              // 0..3 : 32-row group
  const int wc = wave & 1;               // 0..1 : 64-col group
  const int m0 = blockIdx.y * BM;
  const int n0 = blockIdx.x * BN;
  const unsigned asOff = (unsigned)(unsigned long long)(void*)&As[0][0];
  const unsigned wsOff = (unsigned)(unsigned long long)(void*)&Ws[0][0];

  v8f acc[2][4];
  #pragma unroll
  for (int i = 0; i < 2; i++)
    #pragma unroll
    for (int j = 0; j < 4; j++)
      #pragma unroll
      for (int e = 0; e < 8; e++) acc[i][j][e] = 0.f;

  for (int k0 = 0; k0 < K; k0 += BK) {
    // ---- TDM tile staging: wave0 -> A tile, wave1 -> W tile
    if (wave == 0) {
      tdm_load_2d(asOff, A + (size_t)m0 * K + k0, K - k0, M - m0, BK, BM, K);
      __builtin_amdgcn_s_wait_tensorcnt(0);
    } else if (wave == 1) {
      tdm_load_2d(wsOff, W + (size_t)n0 * K + k0, K - k0, N - n0, BK, BN, K);
      __builtin_amdgcn_s_wait_tensorcnt(0);
    }
    __syncthreads();

    const int l2 = lane & 15;
    const int hi = lane >> 4;
    BFrag a[2], b[4];
    #pragma unroll
    for (int rt = 0; rt < 2; rt++) {           // A 16x32: K(e)=hi*8+e(+8 for e>=8)
      int row = wr * 32 + rt * 16 + l2;
      a[rt].h[0] = *(const u16x8*)(&As[row][hi * 8]);
      a[rt].h[1] = *(const u16x8*)(&As[row][hi * 8 + 16]);
    }
    #pragma unroll
    for (int ct = 0; ct < 4; ct++) {           // B 32x16: K(e)=hi*8+e(+8 for e>=8)
      int nn = wc * 64 + ct * 16 + l2;
      b[ct].h[0] = *(const u16x8*)(&Ws[nn][hi * 8]);
      b[ct].h[1] = *(const u16x8*)(&Ws[nn][hi * 8 + 16]);
    }
    #pragma unroll
    for (int rt = 0; rt < 2; rt++)
      #pragma unroll
      for (int ct = 0; ct < 4; ct++)
        acc[rt][ct] = __builtin_amdgcn_wmma_f32_16x16x32_bf16(
            false, a[rt].v, false, b[ct].v, (short)0, acc[rt][ct], false, false);
    __syncthreads();
  }

  // ---- store (C/D layout: lane=N col, vgpr r -> M = hi*8 + r)
  #pragma unroll
  for (int rt = 0; rt < 2; rt++)
    #pragma unroll
    for (int ct = 0; ct < 4; ct++) {
      int n = n0 + wc * 64 + ct * 16 + (lane & 15);
      if (n >= N) continue;
      int mb = m0 + wr * 32 + rt * 16 + (lane >> 4) * 8;
      #pragma unroll
      for (int r = 0; r < 8; r++) {
        float v = acc[rt][ct][r];
        size_t off = (size_t)(mb + r) * N + n;
        if (Cf) Cf[off] = v; else Cb[off] = f2bf(v);
      }
    }
}

// -------------------------------------------- depthwise causal conv4+SiLU
__global__ __launch_bounds__(256)
void k_dwconv_silu(const unsigned short* __restrict__ xz,
                   const float* __restrict__ cw, const float* __restrict__ cb,
                   unsigned short* __restrict__ xc) {
  int bt = blockIdx.x;
  int t  = bt % TSEQ;
  int d  = threadIdx.x;
  float acc = cb[d];
  #pragma unroll
  for (int k = 0; k < 4; k++) {
    int tt = t - 3 + k;
    if (tt >= 0)
      acc += bf2f(xz[((size_t)(bt - t + tt)) * (2 * DINNER) + d]) * cw[d * 4 + k];
  }
  float sv = acc / (1.f + __expf(-acc));
  xc[(size_t)bt * DINNER + d] = f2bf(sv);
}

// -------------------- fused SSM scan: dt-softplus, state, C-dot, D, z-gate
__global__ __launch_bounds__(256)
void k_ssm_scan(const float* __restrict__ dbl, const unsigned short* __restrict__ xc,
                const unsigned short* __restrict__ xz,
                const float* __restrict__ dtw, const float* __restrict__ dtb,
                const float* __restrict__ Alog, const float* __restrict__ Dp,
                unsigned short* __restrict__ y) {
  __shared__ float sbuf[2][GDBL];
  int b = blockIdx.x, d = threadIdx.x;
  float wdt[DTRANK];
  #pragma unroll
  for (int j = 0; j < DTRANK; j++) wdt[j] = dtw[d * DTRANK + j];
  float bdt = dtb[d];
  float Ad[DST];
  #pragma unroll
  for (int s = 0; s < DST; s++) Ad[s] = -__expf(Alog[d * DST + s]);
  float Dd = Dp[d];
  float hst[DST];
  #pragma unroll
  for (int s = 0; s < DST; s++) hst[s] = 0.f;

  for (int t = 0; t < TSEQ; t++) {
    if (d < GDBL) sbuf[t & 1][d] = dbl[((size_t)b * TSEQ + t) * GDBL + d];
    __syncthreads();
    const float* db = sbuf[t & 1];
    float dtr = bdt;
    #pragma unroll
    for (int j = 0; j < DTRANK; j++) dtr += db[j] * wdt[j];
    float dt = dtr > 20.f ? dtr : __logf(1.f + __expf(dtr));
    size_t base = (size_t)b * TSEQ + t;
    float xt = bf2f(xc[base * DINNER + d]);
    float dtx = dt * xt;
    float accv = 0.f;
    #pragma unroll
    for (int s = 0; s < DST; s++) {
      hst[s] = __expf(dt * Ad[s]) * hst[s] + dtx * db[DTRANK + s];
      accv += hst[s] * db[DTRANK + DST + s];
    }
    float z = bf2f(xz[base * (2 * DINNER) + DINNER + d]);
    float g = z / (1.f + __expf(-z));
    y[base * DINNER + d] = f2bf((accv + Dd * xt) * g);
  }
}

// ---------------------------------------------- final LN of last timestep
__global__ __launch_bounds__(32)
void k_final_ln(const float* __restrict__ h, const float* __restrict__ resid,
                const float* __restrict__ w, const float* __restrict__ bia,
                float* __restrict__ comb, int dirOff) {
  int b = blockIdx.x, lane = threadIdx.x;
  size_t base = ((size_t)b * TSEQ + (TSEQ - 1)) * DMOD;
  float r[4]; float s = 0.f;
  #pragma unroll
  for (int j = 0; j < 4; j++) {
    int c = lane + 32 * j;
    float v = h[base + c] + resid[base + c];
    r[j] = v; s += v;
  }
  #pragma unroll
  for (int o = 16; o > 0; o >>= 1) s += __shfl_xor(s, o, 32);
  float mu = s * (1.f / DMOD);
  float vs = 0.f;
  #pragma unroll
  for (int j = 0; j < 4; j++) { float d = r[j] - mu; vs += d * d; }
  #pragma unroll
  for (int o = 16; o > 0; o >>= 1) vs += __shfl_xor(vs, o, 32);
  float rstd = rsqrtf(vs * (1.f / DMOD) + 1e-5f);
  #pragma unroll
  for (int j = 0; j < 4; j++) {
    int c = lane + 32 * j;
    comb[(size_t)b * (2 * DMOD) + dirOff + c] = (r[j] - mu) * rstd * w[c] + bia[c];
  }
}

// --------------------------------------------------------------- head
__global__ __launch_bounds__(256)
void k_head(const float* __restrict__ comb, const float* __restrict__ hw,
            const float* __restrict__ hb, float* __restrict__ out) {
  __shared__ float row[2 * DMOD];
  int b = blockIdx.x, c = threadIdx.x;
  row[c] = comb[(size_t)b * (2 * DMOD) + c];
  __syncthreads();
  float acc = hb[c];
  #pragma unroll 8
  for (int j = 0; j < 2 * DMOD; j++) acc += row[j] * hw[(size_t)c * (2 * DMOD) + j];
  out[(size_t)b * NCLS + c] = acc;
}

// =====================================================================
extern "C" void kernel_launch(void* const* d_in, const int* in_sizes, int n_in,
                              void* d_out, int out_size, void* d_ws, size_t ws_size,
                              hipStream_t stream) {
  (void)in_sizes; (void)n_in; (void)out_size; (void)ws_size;
  const float* x      = (const float*)d_in[0];
  const float* conv_w = (const float*)d_in[1];
  const float* conv_b = (const float*)d_in[2];
  const float* L_inp [2] = {(const float*)d_in[3],  (const float*)d_in[14]};
  const float* L_cw  [2] = {(const float*)d_in[4],  (const float*)d_in[15]};
  const float* L_cb  [2] = {(const float*)d_in[5],  (const float*)d_in[16]};
  const float* L_xp  [2] = {(const float*)d_in[6],  (const float*)d_in[17]};
  const float* L_dtw [2] = {(const float*)d_in[7],  (const float*)d_in[18]};
  const float* L_dtb [2] = {(const float*)d_in[8],  (const float*)d_in[19]};
  const float* L_Alog[2] = {(const float*)d_in[9],  (const float*)d_in[20]};
  const float* L_D   [2] = {(const float*)d_in[10], (const float*)d_in[21]};
  const float* L_outp[2] = {(const float*)d_in[11], (const float*)d_in[22]};
  const float* L_lnw [2] = {(const float*)d_in[12], (const float*)d_in[23]};
  const float* L_lnb [2] = {(const float*)d_in[13], (const float*)d_in[24]};
  const float* nf_w   = (const float*)d_in[25];
  const float* nf_b   = (const float*)d_in[26];
  const float* head_w = (const float*)d_in[27];
  const float* head_b = (const float*)d_in[28];

  char* ws = (char*)d_ws;
  size_t off = 0;
  auto alloc = [&](size_t bytes) -> void* {
    void* p = ws + off;
    off += (bytes + 255) & ~(size_t)255;
    return p;
  };
  const size_t nInp  = (size_t)NLAYER * 2 * DINNER * DMOD;
  const size_t nXp   = (size_t)NLAYER * GDBL * DINNER;
  const size_t nOutp = (size_t)NLAYER * DMOD * DINNER;
  unsigned short* inp_bf [2] = {(unsigned short*)alloc(nInp * 2),  (unsigned short*)alloc(nInp * 2)};
  unsigned short* xp_bf  [2] = {(unsigned short*)alloc(nXp * 2),   (unsigned short*)alloc(nXp * 2)};
  unsigned short* outp_bf[2] = {(unsigned short*)alloc(nOutp * 2), (unsigned short*)alloc(nOutp * 2)};
  float* h0    = (float*)alloc((size_t)NROWS * DMOD * 4);
  float* hA    = (float*)alloc((size_t)NROWS * DMOD * 4);
  float* hB    = (float*)alloc((size_t)NROWS * DMOD * 4);
  float* resid = (float*)alloc((size_t)NROWS * DMOD * 4);
  unsigned short* hn_bf = (unsigned short*)alloc((size_t)NROWS * DMOD * 2);
  unsigned short* xz_bf = (unsigned short*)alloc((size_t)NROWS * 2 * DINNER * 2);
  unsigned short* xc_bf = (unsigned short*)alloc((size_t)NROWS * DINNER * 2);
  unsigned short* y_bf  = (unsigned short*)alloc((size_t)NROWS * DINNER * 2);
  float* dbl  = (float*)alloc((size_t)NROWS * GDBL * 4);
  float* comb = (float*)alloc((size_t)NB * 2 * DMOD * 4);

  // weights -> bf16 (once per launch)
  for (int dir = 0; dir < 2; dir++) {
    k_cvt_bf16<<<256, 256, 0, stream>>>(L_inp[dir],  inp_bf[dir],  (int)nInp);
    k_cvt_bf16<<<64,  256, 0, stream>>>(L_xp[dir],   xp_bf[dir],   (int)nXp);
    k_cvt_bf16<<<128, 256, 0, stream>>>(L_outp[dir], outp_bf[dir], (int)nOutp);
  }

  // stem conv: x(B,L) -> h0(B,T,DM)
  k_conv0<<<NROWS, 128, 0, stream>>>(x, conv_w, conv_b, h0);

  for (int dir = 0; dir < 2; dir++) {
    // (possibly reversed) view of h0
    {
      size_t total = (size_t)NROWS * DMOD;
      k_copy_rev<<<(int)((total + 255) / 256), 256, 0, stream>>>(h0, hA, dir);
    }
    float* hin = hA;
    float* hout = hB;
    for (int l = 0; l < NLAYER; l++) {
      k_ln_res<<<NROWS / 8, 256, 0, stream>>>(hin, resid,
          L_lnw[dir] + l * DMOD, L_lnb[dir] + l * DMOD, hn_bf, l == 0);

      // xz = hn @ inp.T   (M=24576, N=512, K=128) -> bf16
      k_gemm_bf16<<<dim3(2 * 2 * DINNER / BN, NROWS / BM), 256, 0, stream>>>(
          hn_bf, inp_bf[dir] + (size_t)l * 2 * DINNER * DMOD,
          nullptr, xz_bf, NROWS, 2 * DINNER, DMOD);

      k_dwconv_silu<<<NROWS, 256, 0, stream>>>(
          xz_bf, L_cw[dir] + (size_t)l * DINNER * 4, L_cb[dir] + l * DINNER, xc_bf);

      // dbl = xc @ xp.T   (M=24576, N=40, K=256) -> f32 (TDM OOB zero-fill pads N)
      k_gemm_bf16<<<dim3(1, NROWS / BM), 256, 0, stream>>>(
          xc_bf, xp_bf[dir] + (size_t)l * GDBL * DINNER,
          dbl, nullptr, NROWS, GDBL, DINNER);

      k_ssm_scan<<<NB, 256, 0, stream>>>(
          dbl, xc_bf, xz_bf,
          L_dtw[dir] + (size_t)l * DINNER * DTRANK, L_dtb[dir] + l * DINNER,
          L_Alog[dir] + (size_t)l * DINNER * DST, L_D[dir] + l * DINNER, y_bf);

      // h_next = y @ outp.T  (M=24576, N=128, K=256) -> f32
      k_gemm_bf16<<<dim3(1, NROWS / BM), 256, 0, stream>>>(
          y_bf, outp_bf[dir] + (size_t)l * DMOD * DINNER,
          hout, nullptr, NROWS, DMOD, DINNER);

      float* tmp = hin; hin = hout; hout = tmp;
    }
    k_final_ln<<<NB, 32, 0, stream>>>(hin, resid, nf_w, nf_b, comb, dir * DMOD);
  }

  k_head<<<NB, 256, 0, stream>>>(comb, head_w, head_b, (float*)d_out);
}
